// AttentionFreeTransformer_55508157334155
// MI455X (gfx1250) — matrix operand order repaired and checked by reading, not compiled, not run on hardware
//
#include <hip/hip_runtime.h>
#include <stdint.h>

// ---------------------------------------------------------------------------
// Types for CDNA5 WMMA (wave32, 16x16x32 bf16 -> f32)
// ---------------------------------------------------------------------------
typedef __attribute__((ext_vector_type(16))) __bf16 v16bf;
typedef __attribute__((ext_vector_type(8)))  __bf16 v8bf;
typedef __attribute__((ext_vector_type(8)))  float  v8f;

#define EPS_RMS 1.1920929e-07f
#define SCAN_CHUNKS 32
#define SCAN_CLEN   64    // S / SCAN_CHUNKS

static __device__ __forceinline__ unsigned short f32_to_bf16_rne(float f) {
  union { float f; uint32_t u; } cv; cv.f = f;
  uint32_t u = cv.u;
  uint32_t lsb = (u >> 16) & 1u;
  u += 0x7fffu + lsb;            // round to nearest even
  return (unsigned short)(u >> 16);
}

// Combine two 16-byte bf16 chunks into one 16-element WMMA fragment register set
static __device__ __forceinline__ v16bf make_frag(const unsigned short* p0,
                                                  const unsigned short* p1) {
  const v8bf lo = *reinterpret_cast<const v8bf*>(p0);
  const v8bf hi = *reinterpret_cast<const v8bf*>(p1);
  v16bf f;
#pragma unroll
  for (int i = 0; i < 8; ++i) { f[i] = lo[i]; f[8 + i] = hi[i]; }
  return f;
}

// A fragment: 16(M) x 32(K), row m = lane&15 contiguous in K.
//   lane<16  holds K {0..7, 16..23};  lane>=16 holds K {8..15, 24..31}
static __device__ __forceinline__ v16bf load_frag_a(const unsigned short* rowbase_k,
                                                    int half) {
  return make_frag(rowbase_k + half * 8, rowbase_k + half * 8 + 16);
}
// B fragment: 32(K) x 16(N), column n = lane&15 == row n of [N,K] weight (K contig).
//   lane<16 holds K {0..7, 16..23}, lane>=16 holds K {8..15, 24..31}
static __device__ __forceinline__ v16bf load_frag_b(const unsigned short* rowbase_k,
                                                    int half) {
  return make_frag(rowbase_k + half * 8, rowbase_k + half * 8 + 16);
}

static __device__ __forceinline__ v8f wmma_bf16(v16bf a, v16bf b, v8f c) {
  return __builtin_amdgcn_wmma_f32_16x16x32_bf16(
      /*neg_a=*/false, a, /*neg_b=*/false, b,
      /*c_mod=*/(short)0, c, /*reuse_a=*/false, /*reuse_b=*/false);
}

// ---------------------------------------------------------------------------
// Elementwise / reduction kernels
// ---------------------------------------------------------------------------

// fp32 -> bf16 weight conversion, 4 elements / thread (b128 loads)
__global__ void __launch_bounds__(256)
f32_to_bf16_kernel(const float4* __restrict__ in, uint2* __restrict__ out, long n4) {
  long i = (long)blockIdx.x * blockDim.x + threadIdx.x;
  if (i >= n4) return;
  float4 f = in[i];
  unsigned int lo = (unsigned int)f32_to_bf16_rne(f.x) |
                    ((unsigned int)f32_to_bf16_rne(f.y) << 16);
  unsigned int hi = (unsigned int)f32_to_bf16_rne(f.z) |
                    ((unsigned int)f32_to_bf16_rne(f.w) << 16);
  out[i] = make_uint2(lo, hi);
}

// RMS-norm each token row of x (D=2048), emit bf16. One 256-thread block / token.
__global__ void __launch_bounds__(256)
rmsnorm_x_kernel(const float* __restrict__ x, unsigned short* __restrict__ xn) {
  const int D = 2048;
  const size_t t = blockIdx.x;
  const float* row = x + t * D;
  unsigned short* orow = xn + t * D;

  float v[8];
  float ssq = 0.f;
#pragma unroll
  for (int j = 0; j < 8; ++j) {
    float f = row[threadIdx.x + j * 256];
    v[j] = f;
    ssq += f * f;
  }
  __shared__ float red[256];
  red[threadIdx.x] = ssq;
  __syncthreads();
  for (int s = 128; s > 0; s >>= 1) {
    if (threadIdx.x < s) red[threadIdx.x] += red[threadIdx.x + s];
    __syncthreads();
  }
  const float scale = rsqrtf(red[0] / (float)D + EPS_RMS);
#pragma unroll
  for (int j = 0; j < 8; ++j)
    orow[threadIdx.x + j * 256] = f32_to_bf16_rne(v[j] * scale);
}

// Per-token inverse-RMS of q and k slices of qkv [M, 3D]. One block / token.
__global__ void __launch_bounds__(256)
qk_rms_kernel(const float* __restrict__ qkv, float* __restrict__ irq,
              float* __restrict__ irk) {
  const int D = 2048;
  const size_t t = blockIdx.x;
  const float* q = qkv + t * (size_t)(3 * D);
  const float* k = q + D;
  float sq = 0.f, sk = 0.f;
#pragma unroll
  for (int j = 0; j < 8; ++j) {
    float a = q[threadIdx.x + j * 256]; sq += a * a;
    float b = k[threadIdx.x + j * 256]; sk += b * b;
  }
  __shared__ float rq[256], rk[256];
  rq[threadIdx.x] = sq;
  rk[threadIdx.x] = sk;
  __syncthreads();
  for (int s = 128; s > 0; s >>= 1) {
    if (threadIdx.x < s) {
      rq[threadIdx.x] += rq[threadIdx.x + s];
      rk[threadIdx.x] += rk[threadIdx.x + s];
    }
    __syncthreads();
  }
  if (threadIdx.x == 0) {
    irq[t] = rsqrtf(rq[0] / (float)D + EPS_RMS);
    irk[t] = rsqrtf(rk[0] / (float)D + EPS_RMS);
  }
}

// ---------------------------------------------------------------------------
// Blocked causal scan along S (additive prefix): S split into SCAN_CHUNKS
// chunks of SCAN_CLEN. Phase A: per-chunk sums of w and w*v. Phase C: exclusive
// chunk prefix + in-chunk replay emitting y (bf16). 262k threads per phase.
// ---------------------------------------------------------------------------

// Phase A: partial sums per (b, chunk, d).
__global__ void __launch_bounds__(256)
scan_partials_kernel(const float* __restrict__ qkv, const float* __restrict__ irk,
                     float* __restrict__ pw, float* __restrict__ pkv) {
  const int D = 2048, S = 2048;
  const int b = blockIdx.z;
  const int c = blockIdx.y;
  const int d = blockIdx.x * blockDim.x + threadIdx.x;
  float sw = 0.f, skv = 0.f;
  const int s0 = c * SCAN_CLEN;
  for (int s = s0; s < s0 + SCAN_CLEN; ++s) {
    const size_t t = (size_t)b * S + s;
    const float* base = qkv + t * (size_t)(3 * D);
    // pull the next rows toward the WGP while the exp chain executes
    __builtin_prefetch(base + 3 * (size_t)(3 * D) + D + d, 0, 1);
    const float k = base[D + d];
    const float v = base[2 * D + d];
    const float wk = expf(k * irk[t]);
    sw += wk;
    skv = fmaf(wk, v, skv);
  }
  const size_t pi = ((size_t)b * SCAN_CHUNKS + c) * D + d;
  pw[pi]  = sw;
  pkv[pi] = skv;
}

// Phase C: exclusive prefix over chunk partials, then replay chunk -> y.
__global__ void __launch_bounds__(256)
scan_emit_kernel(const float* __restrict__ qkv, const float* __restrict__ irq,
                 const float* __restrict__ irk, const float* __restrict__ pw,
                 const float* __restrict__ pkv, unsigned short* __restrict__ y) {
  const int D = 2048, S = 2048;
  const int b = blockIdx.z;
  const int c = blockIdx.y;
  const int d = blockIdx.x * blockDim.x + threadIdx.x;

  float acc_w = 0.f, acc_kv = 0.f;
  const size_t pbase = (size_t)b * SCAN_CHUNKS * D + d;
  for (int cc = 0; cc < c; ++cc) {          // exclusive chunk prefix (L2 hits)
    acc_w  += pw[pbase + (size_t)cc * D];
    acc_kv += pkv[pbase + (size_t)cc * D];
  }

  const int s0 = c * SCAN_CLEN;
  for (int s = s0; s < s0 + SCAN_CLEN; ++s) {
    const size_t t = (size_t)b * S + s;
    const float* base = qkv + t * (size_t)(3 * D);
    __builtin_prefetch(base + 3 * (size_t)(3 * D) + d, 0, 1);
    const float q = base[d];
    const float k = base[D + d];
    const float v = base[2 * D + d];
    const float wk = expf(k * irk[t]);
    acc_kv = fmaf(wk, v, acc_kv);
    acc_w += wk;
    const float sig = 1.f / (1.f + expf(-q * irq[t]));
    y[t * (size_t)D + d] = f32_to_bf16_rne(sig * (acc_kv / (acc_w + 1e-6f)));
  }
}

// ---------------------------------------------------------------------------
// WMMA GEMM kernels: C[M,N] = A[M,K] @ W[N,K]^T  (A,W bf16)
// Block = 256 threads = 8 waves, wave tile 32x32, block tile 64(M) x 128(N).
// K and LDC are compile-time so the epilogue lowers to immediate-offset store
// clauses and the K-loop strides become constants.
// ---------------------------------------------------------------------------

// GEMM1: qkv(fp32) = xn @ w_qkv^T
template <int K, int LDC>
__global__ void __launch_bounds__(256)
gemm1_kernel(const unsigned short* __restrict__ A, const unsigned short* __restrict__ W,
             float* __restrict__ C) {
  const int lane = threadIdx.x & 31;
  const int wid  = threadIdx.x >> 5;
  const int r = lane & 15, half = lane >> 4;
  const int row0 = blockIdx.x * 64  + (wid >> 2) * 32;
  const int col0 = blockIdx.y * 128 + (wid & 3) * 32;

  const unsigned short* a0 = A + (size_t)(row0 + r)      * K;
  const unsigned short* a1 = A + (size_t)(row0 + 16 + r) * K;
  const unsigned short* b0 = W + (size_t)(col0 + r)      * K;
  const unsigned short* b1 = W + (size_t)(col0 + 16 + r) * K;

  v8f zero = {};
  v8f acc[2][2];
  acc[0][0] = zero; acc[0][1] = zero; acc[1][0] = zero; acc[1][1] = zero;

  for (int k = 0; k < K; k += 32) {
    v16bf af[2], bf2[2];
    af[0]  = load_frag_a(a0 + k, half);
    af[1]  = load_frag_a(a1 + k, half);
    bf2[0] = load_frag_b(b0 + k, half);
    bf2[1] = load_frag_b(b1 + k, half);
#pragma unroll
    for (int i = 0; i < 2; ++i)
#pragma unroll
      for (int j = 0; j < 2; ++j)
        acc[i][j] = wmma_bf16(af[i], bf2[j], acc[i][j]);
  }

  // Base address per lane; all element offsets are compile-time constants.
  float* cbase = C + (size_t)(row0 + half * 8) * LDC + col0 + r;
#pragma unroll
  for (int i = 0; i < 2; ++i)
#pragma unroll
    for (int j = 0; j < 2; ++j)
#pragma unroll
      for (int e = 0; e < 8; ++e)
        cbase[(size_t)(i * 16 + e) * LDC + j * 16] = acc[i][j][e];
}

// GEMM2 + fused SwiGLU: h = u * silu(g), u/g rows of w_swiglu offset by DFF.
template <int K, int DFF>
__global__ void __launch_bounds__(256)
gemm2_swiglu_kernel(const unsigned short* __restrict__ A,
                    const unsigned short* __restrict__ W,
                    unsigned short* __restrict__ H) {
  const int lane = threadIdx.x & 31;
  const int wid  = threadIdx.x >> 5;
  const int r = lane & 15, half = lane >> 4;
  const int row0 = blockIdx.x * 64  + (wid >> 2) * 32;
  const int col0 = blockIdx.y * 128 + (wid & 3) * 32;

  const unsigned short* a0  = A + (size_t)(row0 + r)            * K;
  const unsigned short* a1  = A + (size_t)(row0 + 16 + r)       * K;
  const unsigned short* bu0 = W + (size_t)(col0 + r)            * K;
  const unsigned short* bu1 = W + (size_t)(col0 + 16 + r)       * K;
  const unsigned short* bg0 = W + (size_t)(DFF + col0 + r)      * K;
  const unsigned short* bg1 = W + (size_t)(DFF + col0 + 16 + r) * K;

  v8f zero = {};
  v8f accu[2][2], accg[2][2];
#pragma unroll
  for (int i = 0; i < 2; ++i)
#pragma unroll
    for (int j = 0; j < 2; ++j) { accu[i][j] = zero; accg[i][j] = zero; }

  for (int k = 0; k < K; k += 32) {
    v16bf af[2], bu[2], bg[2];
    af[0] = load_frag_a(a0 + k, half);
    af[1] = load_frag_a(a1 + k, half);
    bu[0] = load_frag_b(bu0 + k, half);
    bu[1] = load_frag_b(bu1 + k, half);
    bg[0] = load_frag_b(bg0 + k, half);
    bg[1] = load_frag_b(bg1 + k, half);
#pragma unroll
    for (int i = 0; i < 2; ++i)
#pragma unroll
      for (int j = 0; j < 2; ++j) {
        accu[i][j] = wmma_bf16(af[i], bu[j], accu[i][j]);
        accg[i][j] = wmma_bf16(af[i], bg[j], accg[i][j]);
      }
  }

  unsigned short* hbase = H + (size_t)(row0 + half * 8) * DFF + col0 + r;
#pragma unroll
  for (int i = 0; i < 2; ++i)
#pragma unroll
    for (int j = 0; j < 2; ++j)
#pragma unroll
      for (int e = 0; e < 8; ++e) {
        const float u = accu[i][j][e];
        const float g = accg[i][j][e];
        const float h = u * (g / (1.f + expf(-g)));    // u * silu(g)
        hbase[(size_t)(i * 16 + e) * DFF + j * 16] = f32_to_bf16_rne(h);
      }
}

// GEMM3 + residual: out = x + h @ w_out^T  (fp32 output)
template <int K, int LDC>
__global__ void __launch_bounds__(256)
gemm3_residual_kernel(const unsigned short* __restrict__ A,
                      const unsigned short* __restrict__ W,
                      const float* __restrict__ x, float* __restrict__ out) {
  const int lane = threadIdx.x & 31;
  const int wid  = threadIdx.x >> 5;
  const int r = lane & 15, half = lane >> 4;
  const int row0 = blockIdx.x * 64  + (wid >> 2) * 32;
  const int col0 = blockIdx.y * 128 + (wid & 3) * 32;

  const unsigned short* a0 = A + (size_t)(row0 + r)      * K;
  const unsigned short* a1 = A + (size_t)(row0 + 16 + r) * K;
  const unsigned short* b0 = W + (size_t)(col0 + r)      * K;
  const unsigned short* b1 = W + (size_t)(col0 + 16 + r) * K;

  v8f zero = {};
  v8f acc[2][2];
  acc[0][0] = zero; acc[0][1] = zero; acc[1][0] = zero; acc[1][1] = zero;

  for (int k = 0; k < K; k += 32) {
    v16bf af[2], bf2[2];
    af[0]  = load_frag_a(a0 + k, half);
    af[1]  = load_frag_a(a1 + k, half);
    bf2[0] = load_frag_b(b0 + k, half);
    bf2[1] = load_frag_b(b1 + k, half);
#pragma unroll
    for (int i = 0; i < 2; ++i)
#pragma unroll
      for (int j = 0; j < 2; ++j)
        acc[i][j] = wmma_bf16(af[i], bf2[j], acc[i][j]);
  }

  const size_t base = (size_t)(row0 + half * 8) * LDC + col0 + r;
  const float* xbase = x + base;
  float* obase = out + base;
#pragma unroll
  for (int i = 0; i < 2; ++i)
#pragma unroll
    for (int j = 0; j < 2; ++j)
#pragma unroll
      for (int e = 0; e < 8; ++e) {
        const size_t off = (size_t)(i * 16 + e) * LDC + j * 16;
        obase[off] = xbase[off] + acc[i][j][e];
      }
}

// ---------------------------------------------------------------------------
// Host launcher
// ---------------------------------------------------------------------------
extern "C" void kernel_launch(void* const* d_in, const int* in_sizes, int n_in,
                              void* d_out, int out_size, void* d_ws, size_t ws_size,
                              hipStream_t stream) {
  (void)in_sizes; (void)n_in; (void)out_size; (void)ws_size;

  const float* x        = (const float*)d_in[0];   // [4,2048,2048]
  const float* w_qkv    = (const float*)d_in[1];   // [6144,2048]
  const float* w_swiglu = (const float*)d_in[2];   // [16384,2048]
  const float* w_out    = (const float*)d_in[3];   // [2048,8192]
  float* out = (float*)d_out;                      // [4,2048,2048]

  const int Bb = 4, S = 2048, D = 2048, DFF = 8192;
  const size_t M = (size_t)Bb * S;                 // 8192 tokens

  // Workspace layout (256B aligned), with aliasing: y reuses xn, h reuses qkv.
  char* ws = (char*)d_ws;
  size_t off = 0;
  auto alloc = [&](size_t bytes) {
    void* p = ws + off;
    off += (bytes + 255) & ~(size_t)255;
    return p;
  };
  unsigned short* wqkv_bf = (unsigned short*)alloc((size_t)3 * D * D * 2);
  unsigned short* wswi_bf = (unsigned short*)alloc((size_t)2 * DFF * D * 2);
  unsigned short* wout_bf = (unsigned short*)alloc((size_t)D * DFF * 2);
  unsigned short* xn_bf   = (unsigned short*)alloc(M * (size_t)D * 2);       // -> also y
  float*          qkv     = (float*)alloc(M * (size_t)(3 * D) * 4);          // -> also h
  float*          irq     = (float*)alloc(M * 4);
  float*          irk     = (float*)alloc(M * 4);
  float*          pw      = (float*)alloc((size_t)Bb * SCAN_CHUNKS * D * 4);
  float*          pkv     = (float*)alloc((size_t)Bb * SCAN_CHUNKS * D * 4);
  unsigned short* y_bf = xn_bf;                    // xn dead after GEMM1
  unsigned short* h_bf = (unsigned short*)qkv;     // qkv dead after scan

  // 1) Weights fp32 -> bf16 (so all three fit ~126MB, mostly resident in 192MB L2)
  {
    const long n4_qkv = (long)3 * D * D / 4;
    const long n4_swi = (long)2 * DFF * D / 4;
    const long n4_out = (long)D * DFF / 4;
    f32_to_bf16_kernel<<<dim3((unsigned)((n4_qkv + 255) / 256)), 256, 0, stream>>>(
        (const float4*)w_qkv, (uint2*)wqkv_bf, n4_qkv);
    f32_to_bf16_kernel<<<dim3((unsigned)((n4_swi + 255) / 256)), 256, 0, stream>>>(
        (const float4*)w_swiglu, (uint2*)wswi_bf, n4_swi);
    f32_to_bf16_kernel<<<dim3((unsigned)((n4_out + 255) / 256)), 256, 0, stream>>>(
        (const float4*)w_out, (uint2*)wout_bf, n4_out);
  }

  // 2) xn = rms(x) -> bf16
  rmsnorm_x_kernel<<<dim3((unsigned)M), 256, 0, stream>>>(x, xn_bf);

  // 3) GEMM1: qkv = xn @ w_qkv^T   [8192 x 6144]
  gemm1_kernel<2048, 6144><<<dim3((unsigned)(M / 64), 3 * D / 128), 256, 0, stream>>>(
      xn_bf, wqkv_bf, qkv);

  // 4) per-token 1/rms for q and k
  qk_rms_kernel<<<dim3((unsigned)M), 256, 0, stream>>>(qkv, irq, irk);

  // 5) blocked causal scan -> y (bf16): partials, then prefix+emit
  scan_partials_kernel<<<dim3(D / 256, SCAN_CHUNKS, Bb), 256, 0, stream>>>(
      qkv, irk, pw, pkv);
  scan_emit_kernel<<<dim3(D / 256, SCAN_CHUNKS, Bb), 256, 0, stream>>>(
      qkv, irq, irk, pw, pkv, y_bf);

  // 6) GEMM2 fused SwiGLU: h = u * silu(g)   [8192 x 8192] bf16
  gemm2_swiglu_kernel<2048, 8192><<<dim3((unsigned)(M / 64), DFF / 128), 256, 0, stream>>>(
      y_bf, wswi_bf, h_bf);

  // 7) GEMM3 + residual: out = x + h @ w_out^T   [8192 x 2048] fp32
  gemm3_residual_kernel<8192, 2048><<<dim3((unsigned)(M / 64), D / 128), 256, 0, stream>>>(
      h_bf, wout_bf, x, out);
}
